// CLGNN_Model_36773509988809
// MI455X (gfx1250) — matrix operand order repaired and checked by reading, not compiled, not run on hardware
//
#include <hip/hip_runtime.h>
#include <hip/hip_bf16.h>
#include <math.h>

// ---------------- problem constants (match reference) ----------------
#define NNODE   100000
#define NFEAT   512
#define NLABEL  64
#define NHID    256
#define DIN     (NFEAT + NLABEL)   // 576

typedef __attribute__((ext_vector_type(16))) __bf16 v16bf;
typedef __attribute__((ext_vector_type(8)))  float  v8f;

__device__ __forceinline__ unsigned short f2bf(float f) {
  unsigned int u = __float_as_uint(f);
  unsigned int r = u + 0x7FFFu + ((u >> 16) & 1u);   // round-to-nearest-even
  return (unsigned short)(r >> 16);
}

// CDNA5 async memory->LDS staging (ASYNCcnt-tracked); per-lane 16B transfer.
__device__ __forceinline__ void async_ld_b128(unsigned lds_byte_addr, const void* gaddr) {
  asm volatile("global_load_async_to_lds_b128 %0, %1, off"
               :: "v"(lds_byte_addr), "v"(gaddr) : "memory");
}
__device__ __forceinline__ void wait_asynccnt0() {
  asm volatile("s_wait_asynccnt 0x0" ::: "memory");
}

// ---------------- small elementwise kernels ----------------
__global__ void k_lab_init(int* lab, int n) {
  int i = blockIdx.x * blockDim.x + threadIdx.x;
  if (i < n) lab[i] = -1;
}

__global__ void k_lab_scatter(const int* __restrict__ idxl, const int* __restrict__ y,
                              int* lab, int L) {
  int i = blockIdx.x * blockDim.x + threadIdx.x;
  if (i < L) { int node = idxl[i]; lab[node] = y[node]; }
}

// h0 = [bf16(x) | one_hot(lab)]  ->  [N, 576] bf16
__global__ void k_build_h0(const float* __restrict__ x, const int* __restrict__ lab,
                           unsigned short* __restrict__ h0, int n) {
  long i = (long)blockIdx.x * blockDim.x + threadIdx.x;
  long total = (long)n * DIN;
  if (i >= total) return;
  int node = (int)(i / DIN), c = (int)(i % DIN);
  float v = (c < NFEAT) ? x[(long)node * NFEAT + c]
                        : ((lab[node] == (c - NFEAT)) ? 1.0f : 0.0f);
  h0[i] = f2bf(v);
}

// Wt[n*K + k] = bf16(W[k*Nn + n])   (transpose + convert, weights are tiny)
__global__ void k_wtrans(const float* __restrict__ W, unsigned short* __restrict__ Wt,
                         int K, int Nn) {
  long i = (long)blockIdx.x * blockDim.x + threadIdx.x;
  long total = (long)K * Nn;
  if (i >= total) return;
  int nn = (int)(i / K), k = (int)(i % K);
  Wt[i] = f2bf(W[(long)k * Nn + nn]);
}

__global__ void k_deg_init(float* deg, int n) {     // self-loop weight 2.0
  int i = blockIdx.x * blockDim.x + threadIdx.x;
  if (i < n) deg[i] = 2.0f;
}

__global__ void k_deg_edges(const int* __restrict__ col, float* deg, int E) {
  int e = blockIdx.x * blockDim.x + threadIdx.x;
  if (e < E) atomicAdd(&deg[col[e]], 1.0f);
}

__global__ void k_dinv(const float* __restrict__ deg, float* dinv, int n) {
  int i = blockIdx.x * blockDim.x + threadIdx.x;
  if (i < n) { float d = deg[i]; dinv[i] = (d > 0.0f) ? rsqrtf(d) : 0.0f; }
}

__global__ void k_zero_f32(float* p, long total) {
  long i = (long)blockIdx.x * blockDim.x + threadIdx.x;
  if (i < total) p[i] = 0.0f;
}

// one wave per edge; lane covers C/32 features; acc[col] += Z[row] * dinv[r]*dinv[c]
__global__ void k_aggregate(const int* __restrict__ row, const int* __restrict__ col,
                            const float* __restrict__ dinv, const float* __restrict__ Z,
                            float* __restrict__ acc, int E, int C) {
  int e = blockIdx.x * (blockDim.x >> 5) + (threadIdx.x >> 5);
  if (e >= E) return;
  int lane = threadIdx.x & 31;
  int s = row[e], t = col[e];
  float w = dinv[s] * dinv[t];
  const float* zs = Z + (long)s * C;
  float* at = acc + (long)t * C;
  for (int c = lane; c < C; c += 32) atomicAdd(&at[c], zs[c] * w);
}

// h_next = bf16(relu(acc + Z * 2*dinv^2 + b))   (self-loop ew=2 folded in)
__global__ void k_gcn_finalize(const float* __restrict__ acc, const float* __restrict__ Z,
                               const float* __restrict__ dinv, const float* __restrict__ b,
                               unsigned short* __restrict__ hout, int n, int C) {
  long i = (long)blockIdx.x * blockDim.x + threadIdx.x;
  long total = (long)n * C;
  if (i >= total) return;
  int node = (int)(i / C), c = (int)(i % C);
  float d = dinv[node];
  float v = acc[i] + Z[i] * (2.0f * d * d) + b[c];
  hout[i] = f2bf(v > 0.0f ? v : 0.0f);
}

// h = bf16(elu(Z + b))
__global__ void k_bias_elu(const float* __restrict__ Z, const float* __restrict__ b,
                           unsigned short* __restrict__ hout, int n, int C) {
  long i = (long)blockIdx.x * blockDim.x + threadIdx.x;
  long total = (long)n * C;
  if (i >= total) return;
  int c = (int)(i % C);
  float v = Z[i] + b[c];
  hout[i] = f2bf(v > 0.0f ? v : (__expf(v) - 1.0f));
}

// one wave per row of 64 logits: out = (Z+b) - logsumexp(Z+b)
__global__ void k_logsoftmax(const float* __restrict__ Z, const float* __restrict__ b,
                             float* __restrict__ out, int n) {
  int r = blockIdx.x * (blockDim.x >> 5) + (threadIdx.x >> 5);
  if (r >= n) return;
  int lane = threadIdx.x & 31;
  long base = (long)r * NLABEL;
  float v0 = Z[base + lane]      + b[lane];
  float v1 = Z[base + lane + 32] + b[lane + 32];
  float m = fmaxf(v0, v1);
  for (int o = 16; o > 0; o >>= 1) m = fmaxf(m, __shfl_xor(m, o, 32));
  float s = __expf(v0 - m) + __expf(v1 - m);
  for (int o = 16; o > 0; o >>= 1) s += __shfl_xor(s, o, 32);
  float lse = m + __logf(s);
  out[base + lane]      = v0 - lse;
  out[base + lane + 32] = v1 - lse;
}

// ---------------- WMMA bf16 GEMM: C[M,Nn] = A[M,K] * B[K,Nn] ----------------
// A: bf16 row-major [M,K];  Bt: bf16 [Nn,K] (pre-transposed weight);  C: f32.
// Block = 256 threads = 8 waves; tile 128(M) x 64(N) x 32(K). Tiles are staged
// into LDS with CDNA5 GLOBAL_LOAD_ASYNC_TO_LDS_B128 (ASYNCcnt), double-buffered,
// and stored in WMMA *fragment order* so each lane reads one contiguous 32B chunk.
// All fragments are fetched before the WMMA chain so the DS clause drains once
// and the four v_wmma issue back-to-back into the matrix pipe.
#define TM 128
#define TN 64
#define TK 32

__global__ __launch_bounds__(256) void k_gemm_bf16_wmma(
    const unsigned short* __restrict__ A,
    const unsigned short* __restrict__ Bt,
    float* __restrict__ C,
    int M, int K, int Nn) {
  __shared__ __align__(32) unsigned short ldsA[2][TM * TK];  // 8 frags x 512 elems
  __shared__ __align__(32) unsigned short ldsB[2][TK * TN];  // 4 frags x 512 elems

  const int tid  = threadIdx.x;
  const int wave = tid >> 5;
  const int lane = tid & 31;
  const int m0   = blockIdx.x * TM;
  const int n0   = blockIdx.y * TN;

  // A loader: thread -> (row 0..127, 16-elem k-segment 0..1); two 16B chunks go to
  // (lane m, j=seg*8..) and (lane m+16, j=seg*8..) of the row-group's fragment.
  const int  a_row    = tid >> 1;
  const int  a_seg    = tid & 1;
  const int  a_m      = a_row & 15;
  const int  a_grp    = a_row >> 4;
  const int  a_off_lo = a_grp * 512 + a_m * 16 + a_seg * 8;        // elems
  const int  a_off_hi = a_grp * 512 + (a_m + 16) * 16 + a_seg * 8; // elems
  int a_rowg = m0 + a_row;
  if (a_rowg > M - 1) a_rowg = M - 1;   // clamp: OOB tiles are never stored
  const long a_gbase  = (long)a_rowg * K + a_seg * 16;

  // B loader: thread -> (col 0..63, 8-elem k-segment 0..3); contiguous in Bt.
  const int  b_n     = tid & 63;
  const int  b_k0    = (tid >> 6) * 8;
  const int  b_sub   = b_n >> 4;
  const int  b_lane  = ((b_k0 & 8) ? 16 : 0) + (b_n & 15);
  const int  b_off   = b_sub * 512 + b_lane * 16 + ((b_k0 >> 1) & 8);     // elems
  const long b_gbase = (long)(n0 + b_n) * K + b_k0;

  // LDS byte addresses for async DMA (flat LDS pointers keep offset in [31:0])
  const unsigned la_lo[2] = {
      (unsigned)(uintptr_t)&ldsA[0][a_off_lo], (unsigned)(uintptr_t)&ldsA[1][a_off_lo] };
  const unsigned la_hi[2] = {
      (unsigned)(uintptr_t)&ldsA[0][a_off_hi], (unsigned)(uintptr_t)&ldsA[1][a_off_hi] };
  const unsigned lb[2] = {
      (unsigned)(uintptr_t)&ldsB[0][b_off],    (unsigned)(uintptr_t)&ldsB[1][b_off] };

  v8f zero = {0.f, 0.f, 0.f, 0.f, 0.f, 0.f, 0.f, 0.f};
  v8f acc[4] = {zero, zero, zero, zero};

  const int nk = K / TK;

  auto load_tile = [&](int buf, int kblk) {
    const unsigned short* ga = A + a_gbase + kblk;
    async_ld_b128(la_lo[buf], ga);          // k  .. k+7
    async_ld_b128(la_hi[buf], ga + 8);      // k+8.. k+15
    async_ld_b128(lb[buf], Bt + b_gbase + kblk);
  };

  load_tile(0, 0);
  wait_asynccnt0();
  __syncthreads();

  for (int kt = 0; kt < nk; ++kt) {
    const int buf = kt & 1;
    if (kt + 1 < nk) load_tile(buf ^ 1, (kt + 1) * TK);  // async fill next buffer

    // Fetch every fragment first (one DS clause, one drain), then chain WMMAs.
    const v16bf af  = *reinterpret_cast<const v16bf*>(&ldsA[buf][wave * 512 + lane * 16]);
    v16bf bf[4];
#pragma unroll
    for (int s = 0; s < 4; ++s)
      bf[s] = *reinterpret_cast<const v16bf*>(&ldsB[buf][s * 512 + lane * 16]);
#pragma unroll
    for (int s = 0; s < 4; ++s)
      acc[s] = __builtin_amdgcn_wmma_f32_16x16x32_bf16(
          false, af, false, bf[s], (short)0, acc[s], false, false);

    wait_asynccnt0();   // next-tile DMA landed in LDS
    __syncthreads();    // visible to all waves; prior reads of next buffer done
  }

  // C/D layout: VGPR r, lane l -> row = r + 8*(l>>4), col = l&15
  const int c_half = lane >> 4;
  const int c_col  = lane & 15;
#pragma unroll
  for (int s = 0; s < 4; ++s) {
#pragma unroll
    for (int r = 0; r < 8; ++r) {
      int rowg = m0 + wave * 16 + r + 8 * c_half;
      if (rowg < M) C[(long)rowg * Nn + n0 + s * 16 + c_col] = acc[s][r];
    }
  }
}

// ---------------- orchestration ----------------
static inline int cdiv(long a, long b) { return (int)((a + b - 1) / b); }

extern "C" void kernel_launch(void* const* d_in, const int* in_sizes, int n_in,
                              void* d_out, int out_size, void* d_ws, size_t ws_size,
                              hipStream_t stream) {
  const float* x    = (const float*)d_in[0];
  const int*   y    = (const int*)d_in[1];
  // d_in[2] predictions: unused by the reference
  const int*   adj  = (const int*)d_in[3];
  const int*   idxl = (const int*)d_in[4];
  // d_in[5] n_sample: single-sample mean == identity
  const float* W0  = (const float*)d_in[6];  const float* b0  = (const float*)d_in[7];
  const float* W1  = (const float*)d_in[8];  const float* b1  = (const float*)d_in[9];
  const float* W2  = (const float*)d_in[10]; const float* b2  = (const float*)d_in[11];
  const float* Wm0 = (const float*)d_in[12]; const float* bm0 = (const float*)d_in[13];
  const float* Wm1 = (const float*)d_in[14]; const float* bm1 = (const float*)d_in[15];
  float* out = (float*)d_out;

  const int n = in_sizes[1];          // N nodes
  const int E = in_sizes[3] / 2;      // edges
  const int L = in_sizes[4];          // labeled count
  const int* row = adj;               // adj[0] = message sources
  const int* col = adj + E;           // adj[1] = aggregation targets

  // workspace carve (256B aligned)
  char* p = (char*)d_ws;
  auto carve = [&](size_t bytes) -> char* {
    char* r = p; p += (bytes + 255) & ~(size_t)255; return r;
  };
  unsigned short* h0   = (unsigned short*)carve((size_t)n * DIN * 2);
  unsigned short* hb   = (unsigned short*)carve((size_t)n * 512 * 2);  // reused activations
  float*          Z    = (float*)carve((size_t)n * 512 * 4);
  float*          acc  = (float*)carve((size_t)n * NHID * 4);
  unsigned short* W0t  = (unsigned short*)carve((size_t)DIN * NHID * 2);
  unsigned short* W1t  = (unsigned short*)carve((size_t)NHID * NHID * 2);
  unsigned short* W2t  = (unsigned short*)carve((size_t)NHID * NHID * 2);
  unsigned short* Wm0t = (unsigned short*)carve((size_t)NHID * 512 * 2);
  unsigned short* Wm1t = (unsigned short*)carve((size_t)512 * NLABEL * 2);
  int*   lab  = (int*)carve((size_t)n * 4);
  float* deg  = (float*)carve((size_t)n * 4);
  float* dinv = (float*)carve((size_t)n * 4);

  const int BT = 256;

  // ---- features: labels + one-hot concat ----
  k_lab_init<<<cdiv(n, BT), BT, 0, stream>>>(lab, n);
  k_lab_scatter<<<cdiv(L, BT), BT, 0, stream>>>(idxl, y, lab, L);
  k_build_h0<<<cdiv((long)n * DIN, BT), BT, 0, stream>>>(x, lab, h0, n);

  // ---- weights -> transposed bf16 ----
  k_wtrans<<<cdiv((long)DIN * NHID, BT), BT, 0, stream>>>(W0, W0t, DIN, NHID);
  k_wtrans<<<cdiv((long)NHID * NHID, BT), BT, 0, stream>>>(W1, W1t, NHID, NHID);
  k_wtrans<<<cdiv((long)NHID * NHID, BT), BT, 0, stream>>>(W2, W2t, NHID, NHID);
  k_wtrans<<<cdiv((long)NHID * 512, BT), BT, 0, stream>>>(Wm0, Wm0t, NHID, 512);
  k_wtrans<<<cdiv((long)512 * NLABEL, BT), BT, 0, stream>>>(Wm1, Wm1t, 512, NLABEL);

  // ---- GCN normalization ----
  k_deg_init<<<cdiv(n, BT), BT, 0, stream>>>(deg, n);
  k_deg_edges<<<cdiv(E, BT), BT, 0, stream>>>(col, deg, E);
  k_dinv<<<cdiv(n, BT), BT, 0, stream>>>(deg, dinv, n);

  // ---- 3 GCN layers ----
  const unsigned short* Wts[3]  = {W0t, W1t, W2t};
  const float*          bias[3] = {b0, b1, b2};
  const int             Kd[3]   = {DIN, NHID, NHID};
  const unsigned short* hin = h0;
  for (int l = 0; l < 3; ++l) {
    dim3 g(cdiv(n, TM), NHID / TN);
    k_gemm_bf16_wmma<<<g, 256, 0, stream>>>(hin, Wts[l], Z, n, Kd[l], NHID);
    k_zero_f32<<<cdiv((long)n * NHID, BT), BT, 0, stream>>>(acc, (long)n * NHID);
    k_aggregate<<<cdiv(E, 8), 256, 0, stream>>>(row, col, dinv, Z, acc, E, NHID);
    k_gcn_finalize<<<cdiv((long)n * NHID, BT), BT, 0, stream>>>(acc, Z, dinv, bias[l], hb, n, NHID);
    hin = hb;
  }

  // ---- MLP head ----
  {
    dim3 g(cdiv(n, TM), 512 / TN);
    k_gemm_bf16_wmma<<<g, 256, 0, stream>>>(hb, Wm0t, Z, n, NHID, 512);
    k_bias_elu<<<cdiv((long)n * 512, BT), BT, 0, stream>>>(Z, bm0, hb, n, 512);
  }
  {
    dim3 g(cdiv(n, TM), NLABEL / TN);
    k_gemm_bf16_wmma<<<g, 256, 0, stream>>>(hb, Wm1t, Z, n, 512, NLABEL);
    k_logsoftmax<<<cdiv(n, 8), 256, 0, stream>>>(Z, bm1, out, n);
  }
  (void)ws_size; (void)out_size; (void)n_in;
}
